// CrossViewSwapAttention_29566554866344
// MI455X (gfx1250) — matrix operand order repaired and checked by reading, not compiled, not run on hardware
//
#include <hip/hip_runtime.h>
#include <hip/hip_bf16.h>

typedef _Float16 f16;
typedef __attribute__((ext_vector_type(16))) _Float16 v16h;
typedef __attribute__((ext_vector_type(8)))  _Float16 v8h;
typedef __attribute__((ext_vector_type(8)))  float    v8f;

#define EPSF   1e-5f
#define SCALEF 0.17677669529663687f  /* 32^-0.5 */
#define NEGBIG (-3.0e4f)

// ---------------------------------------------------------------------------
// WMMA helper (CDNA5 wave32, CDNA5 ISA)
// D = A(16x32 f16) * B(32x16 f16) + C(16x16 f32)
// ---------------------------------------------------------------------------
__device__ __forceinline__ v8f wmma16(v16h a, v16h b, v8f c) {
  return __builtin_amdgcn_wmma_f32_16x16x32_f16(false, a, false, b, (short)0, c,
                                                false, false);
}

// Fragment-ready LDS layouts: operand for lane L is 16 contiguous halves at
//   tileBase + L*16   (32 bytes -> two ds_load_b128)
// Inverse maps used when STAGING elements into LDS:
//   A (16x32): element (m,k): lane = m + 16*((k>>3)&1), half = (k&7)|((k&16)>>1)
//   B (32x16): element (k,n): lane = (n&15) + 16*((k>>4)&1), half = k&15
__device__ __forceinline__ int a_half(int kk) { return (kk & 7) | ((kk & 16) >> 1); }

__device__ __forceinline__ v16h lds_frag(const f16* tileBase) {
  int lane = threadIdx.x & 31;
  return *(const v16h*)(tileBase + lane * 16);
}

// ---------------------------------------------------------------------------
// c_embed[m][o] = sum_j E_inv[m][j][3] * cam_w[o][j]      (6 x 128)
// ---------------------------------------------------------------------------
__global__ void k_cembed(const float* __restrict__ E_inv,
                         const float* __restrict__ cam_w,
                         float* __restrict__ cemb) {
  int idx = blockIdx.x * blockDim.x + threadIdx.x;
  if (idx < 6 * 128) {
    int m = idx >> 7, o = idx & 127;
    const float* E = E_inv + m * 16;
    cemb[idx] = cam_w[o * 4 + 0] * E[3] + cam_w[o * 4 + 1] * E[7] +
                cam_w[o * 4 + 2] * E[11] + cam_w[o * 4 + 3] * E[15];
  }
}

// ---------------------------------------------------------------------------
// img_embed (token-major [16896][128]); one block of 128 threads per position
// ---------------------------------------------------------------------------
__global__ void k_imgembed(const float* __restrict__ I_inv,
                           const float* __restrict__ E_inv,
                           const float* __restrict__ img_w,
                           const float* __restrict__ cemb,
                           float* __restrict__ out) {
  __shared__ float red[128];
  int p = blockIdx.x;           // 0..16895  (n,h,w)
  int o = threadIdx.x;          // channel
  int n = p / 2816, rem = p % 2816;
  int h = rem / 88, w = rem % 88;
  float px = (w * (1.0f / 87.0f)) * 1408.0f;
  float py = (h * (1.0f / 31.0f)) * 512.0f;
  const float* I = I_inv + n * 9;
  float c0 = I[0] * px + I[1] * py + I[2];
  float c1 = I[3] * px + I[4] * py + I[5];
  float c2 = I[6] * px + I[7] * py + I[8];
  const float* E = E_inv + n * 16;
  float d0 = E[0] * c0 + E[1] * c1 + E[2] * c2 + E[3];
  float d1 = E[4] * c0 + E[5] * c1 + E[6] * c2 + E[7];
  float d2 = E[8] * c0 + E[9] * c1 + E[10] * c2 + E[11];
  float d3 = E[12] * c0 + E[13] * c1 + E[14] * c2 + E[15];
  float de = img_w[o * 4 + 0] * d0 + img_w[o * 4 + 1] * d1 +
             img_w[o * 4 + 2] * d2 + img_w[o * 4 + 3] * d3 -
             cemb[n * 128 + o];
  red[o] = de * de;
  __syncthreads();
  for (int s = 64; s > 0; s >>= 1) {
    if (o < s) red[o] += red[o + s];
    __syncthreads();
  }
  float inv = 1.0f / fmaxf(sqrtf(red[0]), 1e-12f);
  out[(size_t)p * 128 + o] = de * inv;
}

// ---------------------------------------------------------------------------
// q_in[(n*16384+t)][o] = l2n(w_embed - c_embed[n])[o] + x[o][t]
// ---------------------------------------------------------------------------
__global__ void k_qbuild(const float* __restrict__ bev_grid,
                         const float* __restrict__ bev_w,
                         const float* __restrict__ bev_b,
                         const float* __restrict__ cemb,
                         const float* __restrict__ x,
                         float* __restrict__ qin) {
  __shared__ float red[128];
  int p = blockIdx.x;           // 0..98303
  int o = threadIdx.x;
  int n = p / 16384, pos = p % 16384;
  float we = bev_w[o * 2 + 0] * bev_grid[pos] +
             bev_w[o * 2 + 1] * bev_grid[16384 + pos] + bev_b[o];
  float v = we - cemb[n * 128 + o];
  red[o] = v * v;
  __syncthreads();
  for (int s = 64; s > 0; s >>= 1) {
    if (o < s) red[o] += red[o + s];
    __syncthreads();
  }
  float inv = 1.0f / fmaxf(sqrtf(red[0]), 1e-12f);
  qin[(size_t)p * 128 + o] = v * inv + x[(size_t)o * 16384 + pos];
}

// ---------------------------------------------------------------------------
// Generic fused (LN | BN-ReLU | none) -> f16 -> WMMA GEMM -> (+bias, GELU,
// residual) -> f32/f16 out.  Templated on (KIN, NOUT); A-fragments stay
// register-resident; B loads pipelined; outer loops kept rolled for code size.
// 256 threads / 8 waves; 128 rows per block.
// ---------------------------------------------------------------------------
__device__ __forceinline__ float ldx(const float* X32, const f16* X16, int hw,
                                     int Kin, int t, int c) {
  if (X16) return (float)X16[(size_t)t * Kin + c];
  if (hw > 0) {
    int cam = t / hw;
    int p = t - cam * hw;
    return X32[((size_t)cam * Kin + c) * hw + p];
  }
  return X32[(size_t)t * Kin + c];
}

template <int KIN, int NOUT>
__global__ void k_ln_gemm(const float* __restrict__ X32,
                          const f16* __restrict__ X16, int hw, int pre,
                          const float* __restrict__ gg,
                          const float* __restrict__ bb,
                          const float* __restrict__ W, int wtrans,
                          const float* __restrict__ bias, int act,
                          const float* __restrict__ res, int res_hw,
                          float* __restrict__ O32, f16* __restrict__ O16) {
  constexpr int KT = KIN >> 5;    // K tiles (4 or 8)
  constexpr int NT = NOUT >> 4;   // N tiles (8 or 16)
  extern __shared__ char smem[];
  f16* sWf = (f16*)smem;                 // [KIN*NOUT]  B-fragment order
  f16* sAf = sWf + KIN * NOUT;           // [128*KIN]   A-fragment order
  float* sBias = (float*)(sAf + 128 * KIN);
  const int tid = threadIdx.x, wv = tid >> 5, lane = tid & 31;

  // ---- stage W in B-fragment order: one v16h store per 16 elements ----
  constexpr int WTASKS = (KIN * NOUT) >> 4;
#pragma unroll 1
  for (int idx = tid; idx < WTASKS; idx += 256) {
    int laneT = idx & 31;
    int tile = idx >> 5;
    int kc = tile / NT, nt = tile - kc * NT;
    int g = laneT >> 4, nn = laneT & 15;
    int n = nt * 16 + nn;
    int kbase = kc * 32 + 16 * g;
    v16h tmp;
#pragma unroll
    for (int i = 0; i < 16; ++i) {
      int k = kbase + i;
      float wvv = wtrans ? W[(size_t)n * KIN + k] : W[(size_t)k * NOUT + n];
      tmp[i] = (f16)wvv;
    }
    *(v16h*)(sWf + (size_t)(tile * 32 + laneT) * 16) = tmp;
  }
#pragma unroll 1
  for (int n = tid; n < NOUT; n += 256) sBias[n] = bias ? bias[n] : 0.0f;

  const int tb = blockIdx.x * 128 + wv * 16;
  const float bnscale = rsqrtf(1.0f + EPSF);
  constexpr int PER = KIN >> 5;   // channels per lane (4 or 8)
  const int ahalf = lane & 15;
  const int alane16 = 16 * (lane >> 4);

#pragma unroll 1
  for (int r = 0; r < 16; ++r) {
    int t = tb + r;
    float v[PER];
    float s1 = 0.0f, s2 = 0.0f;
#pragma unroll
    for (int e = 0; e < PER; ++e) {
      int c = lane + (e << 5);
      float xv = ldx(X32, X16, hw, KIN, t, c);
      v[e] = xv;
      s1 += xv;
      s2 += xv * xv;
    }
    if (pre == 1) {
      for (int d = 1; d < 32; d <<= 1) {
        s1 += __shfl_xor(s1, d, 32);
        s2 += __shfl_xor(s2, d, 32);
      }
      float mean = s1 * (1.0f / KIN);
      float var = s2 * (1.0f / KIN) - mean * mean;
      float rstd = rsqrtf(var + EPSF);
#pragma unroll
      for (int e = 0; e < PER; ++e) {
        int c = lane + (e << 5);
        v[e] = (v[e] - mean) * rstd * gg[c] + bb[c];
      }
    } else if (pre == 2) {
#pragma unroll
      for (int e = 0; e < PER; ++e) {
        int c = lane + (e << 5);
        v[e] = fmaxf(v[e] * bnscale * gg[c] + bb[c], 0.0f);
      }
    }
    // scatter into A-fragment order: c = lane + 32*e -> tile (wv*KT+e)
    int laneA = r + alane16;
#pragma unroll
    for (int e = 0; e < PER; ++e) {
      sAf[(size_t)((wv * KT + e) * 32 + laneA) * 16 + ahalf] = (f16)v[e];
    }
  }
  __syncthreads();

  // ---- A fragments register-resident (reused across all N tiles) ----
  v16h afr[KT];
#pragma unroll
  for (int kc = 0; kc < KT; ++kc)
    afr[kc] = lds_frag(sAf + (size_t)((wv * KT + kc) * 32) * 16);

  const int nl = lane & 15, g = lane >> 4;
#pragma unroll 1
  for (int nt = 0; nt < NT; ++nt) {
    v8f acc = {0.f, 0.f, 0.f, 0.f, 0.f, 0.f, 0.f, 0.f};
    // software-pipelined B loads: fetch kc+1 before WMMA on kc
    v16h b = lds_frag(sWf + (size_t)((0 * NT + nt) * 32) * 16);
#pragma unroll
    for (int kc = 0; kc < KT; ++kc) {
      int kn = (kc + 1 < KT) ? kc + 1 : kc;
      v16h bnext = lds_frag(sWf + (size_t)((kn * NT + nt) * 32) * 16);
      acc = wmma16(afr[kc], b, acc);
      b = bnext;
    }
    int n = nt * 16 + nl;
    float bsv = sBias[n];
#pragma unroll
    for (int r = 0; r < 8; ++r) {
      int t = tb + 8 * g + r;
      float o = acc[r] + bsv;
      if (act == 1) o = 0.5f * o * (1.0f + erff(o * 0.70710678f));
      if (res)
        o += (res_hw > 0) ? res[(size_t)n * res_hw + t]
                          : res[(size_t)t * NOUT + n];
      if (O32) O32[(size_t)t * NOUT + n] = o;
      if (O16) O16[(size_t)t * NOUT + n] = (f16)o;
    }
  }
}

// ---------------------------------------------------------------------------
// Windowed cross attention.  grid = (2 q-halves, 6 cams, 64 windows),
// 256 threads / 8 waves; wave = 16 q rows; heads looped (4 x dim 32).
// Register-frugal two-phase softmax through LDS (A-fragment order, so the
// S round-trip is vectorized b128 traffic).  No accumulator spills.
// ---------------------------------------------------------------------------
__device__ __forceinline__ size_t win_ktok(int kr, int Xw, int Yw, int shifted) {
  int kn = kr / 44, rem = kr % 44;
  int ki = rem / 11, kj = rem % 11;
  int kh = shifted ? (ki * 8 + Xw) : (Xw * 4 + ki);
  int kw = shifted ? (kj * 8 + Yw) : (Yw * 11 + kj);
  return (size_t)kn * 2816 + kh * 88 + kw;
}

__global__ void k_attn(const f16* __restrict__ qf, const f16* __restrict__ kf,
                       const f16* __restrict__ vf, f16* __restrict__ aout,
                       int shifted, int qbroad) {
  extern __shared__ char smem[];
  f16* sKf = (f16*)smem;               // 17 tiles * 512 halves (B frag for S)
  f16* sVf = sKf + 17 * 512;           // 18 tiles * 512 halves (B frag for PV)
  f16* sP  = sVf + 18 * 512;           // per wave: 9 tiles * 512 (A frag)
  float* sStat = (float*)(sP + 8 * 9 * 512);  // per wave: 16 mx + 16 inv
  const int tid = threadIdx.x, wv = tid >> 5, lane = tid & 31;
  const int Xw = blockIdx.z >> 3, Yw = blockIdx.z & 7;
  const int cam = blockIdx.y;
  const int qbase = blockIdx.x * 128 + wv * 16;  // in-window q row base
  f16* sPw = sP + (size_t)wv * (9 * 512);
  float* sMxW = sStat + wv * 32;
  float* sInvW = sMxW + 16;
  const int nl = lane & 15, g = lane >> 4, mrow = 8 * g;

#pragma unroll 1
  for (int head = 0; head < 4; ++head) {
    // ---- stage K: tile j, lane = (kr&15)+16*g holds 16 contiguous channels
    // ch = head*32 + 16*g + i  -> one 32B contiguous global read per task ----
#pragma unroll 1
    for (int idx = tid; idx < 17 * 32; idx += 256) {
      int laneT = idx & 31, j = idx >> 5;
      int gT = laneT >> 4;
      int kr = j * 16 + (laneT & 15);
      v16h tmp = {};
      if (kr < 264) {
        size_t ktok = win_ktok(kr, Xw, Yw, shifted);
        tmp = *(const v16h*)(kf + ktok * 128 + head * 32 + 16 * gT);
      }
      *(v16h*)(sKf + (size_t)(j * 32 + laneT) * 16) = tmp;
    }
    // ---- stage V: tile (kc,nt), lane = nn+16*g holds k rows kc*32+8*g+(i&7)+16*(i>>3) ----
#pragma unroll 1
    for (int idx = tid; idx < 18 * 32; idx += 256) {
      int laneT = idx & 31, tile = idx >> 5;
      int kc = tile >> 1, nt = tile & 1;
      int gT = laneT >> 4, nn = laneT & 15;
      int ch = head * 32 + nt * 16 + nn;
      int kbase = kc * 32 + 8 * gT;
      v16h tmp;
#pragma unroll
      for (int i = 0; i < 16; ++i) {
        int kr = kbase + (i & 7) + ((i & 8) << 1);
        f16 vvv = (f16)0.0f;
        if (kr < 264) {
          size_t ktok = win_ktok(kr, Xw, Yw, shifted);
          vvv = vf[ktok * 128 + ch];
        }
        tmp[i] = vvv;
      }
      *(v16h*)(sVf + (size_t)(tile * 32 + laneT) * 16) = tmp;
    }
    // prefetch next head's K/V rows to cache (global_prefetch_b8)
    if (head < 3) {
      int kr = tid & 255;
      if (kr < 264) {
        size_t ktok = win_ktok(kr, Xw, Yw, shifted);
        __builtin_prefetch(kf + ktok * 128 + (head + 1) * 32, 0, 0);
        __builtin_prefetch(vf + ktok * 128 + (head + 1) * 32, 0, 0);
      }
    }
    __syncthreads();

    // ---- Q fragment: two contiguous v8h global loads ----
    int qi = qbase + nl;
    int qh = Xw * 16 + (qi >> 4), qw = Yw * 16 + (qi & 15);
    size_t qt = (size_t)(qbroad ? 0 : cam * 16384) + qh * 128 + qw;
    const f16* qp = qf + qt * 128 + head * 32;
    v8h qlo = *(const v8h*)(qp + 16 * g);
    v8h qhi = *(const v8h*)(qp + 16 * g + 8);
    v16h aq;
#pragma unroll
    for (int i = 0; i < 8; ++i) { aq[i] = qlo[i]; aq[8 + i] = qhi[i]; }

    // ---- phase 1: S blocks -> sPw (A-frag order), track row max ----
    float mx[8];
#pragma unroll
    for (int r = 0; r < 8; ++r) mx[r] = NEGBIG;
    {
      v16h bk = lds_frag(sKf);
#pragma unroll 1
      for (int j = 0; j < 17; ++j) {
        int jn = (j + 1 < 17) ? j + 1 : j;
        v16h bkn = lds_frag(sKf + (size_t)(jn * 32) * 16);
        v8f z8 = {0.f, 0.f, 0.f, 0.f, 0.f, 0.f, 0.f, 0.f};
        v8f s8 = wmma16(aq, bk, z8);
        bk = bkn;
        int col = j * 16 + nl;
        bool ok = col < 264;
        int kc = col >> 5, kk = col & 31;
        int gA = (kk >> 3) & 1;
        int iA = a_half(kk);
#pragma unroll
        for (int r = 0; r < 8; ++r) {
          float s = ok ? s8[r] * SCALEF : NEGBIG;
          mx[r] = fmaxf(mx[r], s);
          sPw[(size_t)(kc * 32 + (r + mrow) + 16 * gA) * 16 + iA] = (f16)s;
        }
      }
    }
    // pad halves 8..15 of tile kc=8 (cols 272..287) with -3e4 so exp()->0
    {
      v8h pad;
#pragma unroll
      for (int i = 0; i < 8; ++i) pad[i] = (f16)NEGBIG;
      *(v8h*)(sPw + (size_t)(8 * 32 + lane) * 16 + 8) = pad;
    }
    // row max across the 16-lane half; publish per-row to sMxW
#pragma unroll
    for (int r = 0; r < 8; ++r)
      for (int d = 1; d < 16; d <<= 1)
        mx[r] = fmaxf(mx[r], __shfl_xor(mx[r], d, 32));
    if (nl == 0) {
#pragma unroll
      for (int r = 0; r < 8; ++r) sMxW[mrow + r] = mx[r];
    }
    __syncthreads();

    // ---- phase 2: exp in A-layout (vectorized reload), row sums ----
    {
      float mxv = sMxW[nl];   // A-layout: this lane holds row nl
      float rsum = 0.0f;
#pragma unroll 1
      for (int kc = 0; kc < 9; ++kc) {
        f16* tp = sPw + (size_t)(kc * 32 + lane) * 16;
        v16h pv = *(const v16h*)tp;
        v16h ev;
#pragma unroll
        for (int i = 0; i < 16; ++i) {
          float e = __expf((float)pv[i] - mxv);
          rsum += e;
          ev[i] = (f16)e;
        }
        *(v16h*)tp = ev;
      }
      rsum += __shfl_xor(rsum, 16, 32);   // combine the two column halves
      if (lane < 16) sInvW[lane] = 1.0f / rsum;
    }
    __syncthreads();

    // ---- O = (P V) * (1/sum): pipelined contiguous v16h operand loads ----
    float invr[8];
#pragma unroll
    for (int r = 0; r < 8; ++r) invr[r] = sInvW[mrow + r];
#pragma unroll 1
    for (int nt = 0; nt < 2; ++nt) {
      v8f acco = {0.f, 0.f, 0.f, 0.f, 0.f, 0.f, 0.f, 0.f};
      v16h ap = lds_frag(sPw);
      v16h bv = lds_frag(sVf + (size_t)(nt * 32) * 16);
#pragma unroll
      for (int kc = 0; kc < 9; ++kc) {
        int kn = (kc + 1 < 9) ? kc + 1 : kc;
        v16h apn = lds_frag(sPw + (size_t)(kn * 32) * 16);
        v16h bvn = lds_frag(sVf + (size_t)((kn * 2 + nt) * 32) * 16);
        acco = wmma16(ap, bv, acco);
        ap = apn;
        bv = bvn;
      }
      int ch = head * 32 + nt * 16 + nl;
#pragma unroll
      for (int r = 0; r < 8; ++r) {
        int qrow = qbase + mrow + r;
        int h2 = Xw * 16 + (qrow >> 4), w2 = Yw * 16 + (qrow & 15);
        size_t arow = (size_t)cam * 16384 + h2 * 128 + w2;
        aout[arow * 128 + ch] = (f16)(acco[r] * invr[r]);
      }
    }
    __syncthreads();
  }
}

// ---------------------------------------------------------------------------
// mean over 6 cameras: out[t][c] = (1/6) sum_n a[(n*16384+t)*128+c]
// ---------------------------------------------------------------------------
__global__ void k_mean6(const f16* __restrict__ a, float* __restrict__ o) {
  size_t idx = (size_t)blockIdx.x * 256 + threadIdx.x;  // < 16384*128
  float s = 0.0f;
#pragma unroll
  for (int n = 0; n < 6; ++n) s += (float)a[(size_t)n * 16384 * 128 + idx];
  o[idx] = s * (1.0f / 6.0f);
}

// ---------------------------------------------------------------------------
// final LayerNorm + transpose to channel-first (1,128,128,128)
// ---------------------------------------------------------------------------
__global__ void k_postln(const float* __restrict__ X,
                         const float* __restrict__ g,
                         const float* __restrict__ b,
                         float* __restrict__ out) {
  __shared__ float red[128];
  int t = blockIdx.x, c = threadIdx.x;
  float v = X[(size_t)t * 128 + c];
  red[c] = v;
  __syncthreads();
  for (int s = 64; s > 0; s >>= 1) {
    if (c < s) red[c] += red[c + s];
    __syncthreads();
  }
  float mean = red[0] * (1.0f / 128.0f);
  __syncthreads();
  float d = v - mean;
  red[c] = d * d;
  __syncthreads();
  for (int s = 64; s > 0; s >>= 1) {
    if (c < s) red[c] += red[c + s];
    __syncthreads();
  }
  float rstd = rsqrtf(red[0] * (1.0f / 128.0f) + EPSF);
  out[(size_t)c * 16384 + t] = d * rstd * g[c] + b[c];
}

// ---------------------------------------------------------------------------
// host launcher
// ---------------------------------------------------------------------------
static inline size_t lngemm_shmem(int Kin, int Nout) {
  return (size_t)2 * Kin * Nout + (size_t)2 * 128 * Kin + 4 * (size_t)Nout;
}

extern "C" void kernel_launch(void* const* d_in, const int* in_sizes, int n_in,
                              void* d_out, int out_size, void* d_ws,
                              size_t ws_size, hipStream_t stream) {
  // input order = setup_inputs() dict order, params flattened in dict order
  const float* x       = (const float*)d_in[0];   // (1,128,128,128)
  const float* feature = (const float*)d_in[1];   // (1,6,128,32,88)
  const float* I_inv   = (const float*)d_in[2];   // (1,6,3,3)
  const float* E_inv   = (const float*)d_in[3];   // (1,6,4,4)
  const float* bevgrid = (const float*)d_in[4];   // (3,128,128)
  const float* cam_w   = (const float*)d_in[5];
  const float* img_w   = (const float*)d_in[6];
  const float* bev_w   = (const float*)d_in[7];
  const float* bev_b   = (const float*)d_in[8];
  const float* proj_g  = (const float*)d_in[9];
  const float* proj_b  = (const float*)d_in[10];
  const float* proj_w  = (const float*)d_in[11];
  const float* lin_g   = (const float*)d_in[12];
  const float* lin_b   = (const float*)d_in[13];
  const float* lin_w   = (const float*)d_in[14];
  const float* const* a1 = (const float* const*)&d_in[15];  // attn1: 14 tensors
  const float* const* a2 = (const float* const*)&d_in[29];  // attn2: 14 tensors
  const float* pre1_g = (const float*)d_in[43];
  const float* pre1_b = (const float*)d_in[44];
  const float* m1w1   = (const float*)d_in[45];
  const float* m1b1   = (const float*)d_in[46];
  const float* m1w2   = (const float*)d_in[47];
  const float* m1b2   = (const float*)d_in[48];
  const float* pre2_g = (const float*)d_in[49];
  const float* pre2_b = (const float*)d_in[50];
  const float* m2w1   = (const float*)d_in[51];
  const float* m2b1   = (const float*)d_in[52];
  const float* m2w2   = (const float*)d_in[53];
  const float* m2b2   = (const float*)d_in[54];
  const float* post_g = (const float*)d_in[55];
  const float* post_b = (const float*)d_in[56];
  float* out = (float*)d_out;
  // attn param dict order: lnq_g,lnq_b,wq,bq,lnk_g,lnk_b,wk,bk,lnv_g,lnv_b,wv,bv,wp,bp
  enum { LNQG, LNQB, WQ, BQ, LNKG, LNKB, WK, BK, LNVG, LNVB, WV, BV, WP, BP };

  // workspace carve
  char* cur = (char*)d_ws;
  auto alloc = [&](size_t bytes) -> char* {
    char* p = cur;
    cur += (bytes + 255) & ~(size_t)255;
    return p;
  };
  float* cemb   = (float*)alloc(6 * 128 * 4);
  float* imgemb = (float*)alloc((size_t)16896 * 128 * 4);
  float* key32  = (float*)alloc((size_t)16896 * 128 * 4);
  float* val32  = (float*)alloc((size_t)16896 * 128 * 4);
  float* qin    = (float*)alloc((size_t)98304 * 128 * 4);
  f16*   qfb    = (f16*)alloc((size_t)98304 * 128 * 2);
  f16*   kfb    = (f16*)alloc((size_t)16896 * 128 * 2);
  f16*   vfb    = (f16*)alloc((size_t)16896 * 128 * 2);
  f16*   ab     = (f16*)alloc((size_t)98304 * 128 * 2);
  float* amean  = (float*)alloc((size_t)16384 * 128 * 4);
  float* zb     = (float*)alloc((size_t)16384 * 128 * 4);
  f16*   hid    = (f16*)alloc((size_t)16384 * 256 * 2);
  float* q1     = (float*)alloc((size_t)16384 * 128 * 4);

  hipFuncSetAttribute((const void*)k_ln_gemm<128, 128>,
                      hipFuncAttributeMaxDynamicSharedMemorySize, 160 * 1024);
  hipFuncSetAttribute((const void*)k_ln_gemm<128, 256>,
                      hipFuncAttributeMaxDynamicSharedMemorySize, 160 * 1024);
  hipFuncSetAttribute((const void*)k_ln_gemm<256, 128>,
                      hipFuncAttributeMaxDynamicSharedMemorySize, 160 * 1024);
  hipFuncSetAttribute((const void*)k_attn,
                      hipFuncAttributeMaxDynamicSharedMemorySize, 120 * 1024);

  const size_t sh_attn =
      (size_t)2 * (17 * 512 + 18 * 512 + 8 * 9 * 512) + 8 * 32 * 4;

  // ---- geometry ----
  k_cembed<<<3, 256, 0, stream>>>(E_inv, cam_w, cemb);
  k_imgembed<<<16896, 128, 0, stream>>>(I_inv, E_inv, img_w, cemb, imgemb);
  k_qbuild<<<98304, 128, 0, stream>>>(bevgrid, bev_w, bev_b, cemb, x, qin);

  // ---- key/val 1x1 convs (bn-relu + GEMM, W is (out,in) -> wtrans) ----
  k_ln_gemm<128, 128><<<132, 256, lngemm_shmem(128, 128), stream>>>(
      feature, nullptr, 2816, 2, proj_g, proj_b, proj_w, 1, nullptr, 0, imgemb,
      0, key32, nullptr);
  k_ln_gemm<128, 128><<<132, 256, lngemm_shmem(128, 128), stream>>>(
      feature, nullptr, 2816, 2, lin_g, lin_b, lin_w, 1, nullptr, 0, nullptr,
      0, val32, nullptr);

  // ================= pass 1 (non-shifted k/v windows) =================
  k_ln_gemm<128, 128><<<768, 256, lngemm_shmem(128, 128), stream>>>(
      qin, nullptr, 0, 1, a1[LNQG], a1[LNQB], a1[WQ], 0, a1[BQ], 0, nullptr, 0,
      nullptr, qfb);
  k_ln_gemm<128, 128><<<132, 256, lngemm_shmem(128, 128), stream>>>(
      key32, nullptr, 0, 1, a1[LNKG], a1[LNKB], a1[WK], 0, a1[BK], 0, nullptr,
      0, nullptr, kfb);
  k_ln_gemm<128, 128><<<132, 256, lngemm_shmem(128, 128), stream>>>(
      val32, nullptr, 0, 1, a1[LNVG], a1[LNVB], a1[WV], 0, a1[BV], 0, nullptr,
      0, nullptr, vfb);

  k_attn<<<dim3(2, 6, 64), 256, sh_attn, stream>>>(qfb, kfb, vfb, ab, 0, 0);
  k_mean6<<<8192, 256, 0, stream>>>(ab, amean);

  // z = amean @ wp + bp + skip(x, channel-first)
  k_ln_gemm<128, 128><<<128, 256, lngemm_shmem(128, 128), stream>>>(
      amean, nullptr, 0, 0, nullptr, nullptr, a1[WP], 0, a1[BP], 0, x, 16384,
      zb, nullptr);

  // q1 = z + GELU(LN(z)@W1+b1)@W2+b2
  k_ln_gemm<128, 256><<<128, 256, lngemm_shmem(128, 256), stream>>>(
      zb, nullptr, 0, 1, pre1_g, pre1_b, m1w1, 0, m1b1, 1, nullptr, 0, nullptr,
      hid);
  k_ln_gemm<256, 128><<<128, 256, lngemm_shmem(256, 128), stream>>>(
      nullptr, hid, 0, 0, nullptr, nullptr, m1w2, 0, m1b2, 0, zb, 0, q1,
      nullptr);

  // ================= pass 2 (shifted k/v windows, broadcast q) =========
  k_ln_gemm<128, 128><<<128, 256, lngemm_shmem(128, 128), stream>>>(
      q1, nullptr, 0, 1, a2[LNQG], a2[LNQB], a2[WQ], 0, a2[BQ], 0, nullptr, 0,
      nullptr, qfb);
  k_ln_gemm<128, 128><<<132, 256, lngemm_shmem(128, 128), stream>>>(
      key32, nullptr, 0, 1, a2[LNKG], a2[LNKB], a2[WK], 0, a2[BK], 0, nullptr,
      0, nullptr, kfb);
  k_ln_gemm<128, 128><<<132, 256, lngemm_shmem(128, 128), stream>>>(
      val32, nullptr, 0, 1, a2[LNVG], a2[LNVB], a2[WV], 0, a2[BV], 0, nullptr,
      0, nullptr, vfb);

  k_attn<<<dim3(2, 6, 64), 256, sh_attn, stream>>>(qfb, kfb, vfb, ab, 1, 1);
  k_mean6<<<8192, 256, 0, stream>>>(ab, amean);

  // z2 = amean @ wp2 + bp2 + q1
  k_ln_gemm<128, 128><<<128, 256, lngemm_shmem(128, 128), stream>>>(
      amean, nullptr, 0, 0, nullptr, nullptr, a2[WP], 0, a2[BP], 0, q1, 0, zb,
      nullptr);

  // q3 = z2 + GELU(LN(z2)@W1+b1)@W2+b2   (into q1 buffer)
  k_ln_gemm<128, 256><<<128, 256, lngemm_shmem(128, 256), stream>>>(
      zb, nullptr, 0, 1, pre2_g, pre2_b, m2w1, 0, m2b1, 1, nullptr, 0, nullptr,
      hid);
  k_ln_gemm<256, 128><<<128, 256, lngemm_shmem(256, 128), stream>>>(
      nullptr, hid, 0, 0, nullptr, nullptr, m2w2, 0, m2b2, 0, zb, 0, q1,
      nullptr);

  // final LN + channel-first transpose
  k_postln<<<16384, 128, 0, stream>>>(q1, post_g, post_b, out);

  (void)in_sizes; (void)n_in; (void)out_size; (void)ws_size;
}
